// BertBilstmNerModel_8254927142979
// MI455X (gfx1250) — matrix operand order repaired and checked by reading, not compiled, not run on hardware
//
#include <hip/hip_runtime.h>
#include <hip/hip_bf16.h>

typedef __attribute__((ext_vector_type(16))) _Float16 v16h;
typedef __attribute__((ext_vector_type(8)))  float    v8f;

// Model dims
static const int NB    = 8;     // batch
static const int SQ    = 128;   // seq len
static const int NT    = NB * SQ;   // 1024 tokens
static const int HD    = 768;
static const int NHEAD = 12;
static const int DHEAD = 64;
static const int FFD   = 3072;
static const int NLAY  = 12;
static const int LH    = 256;   // LSTM hidden
static const int G4    = 4 * LH;
static const int NC    = 9;     // classes

// ---------------------------------------------------------------------------
// CDNA5 async copy: 16 bytes global -> LDS, tracked by ASYNCcnt.
// VDST = LDS byte address (VGPR), VADDR = 64-bit global address (VGPR pair).
// ---------------------------------------------------------------------------
__device__ __forceinline__ void async_b128(unsigned int lds_off, const float* g)
{
    asm volatile("global_load_async_to_lds_b128 %0, %1, off"
                 :: "v"(lds_off), "v"(g) : "memory");
}

// ---------------------------------------------------------------------------
// Issue the 6 per-thread async copies for one 128x32 A panel + 32x64 B panel.
// ---------------------------------------------------------------------------
template <int WT>
__device__ __forceinline__ void stage_tiles(float* shA, float* shB,
                                            const float* Ap, const float* Bp,
                                            int tm, int tn, int k0,
                                            int lda, int ldb, int tid)
{
    // A panel (128 rows x 32 k) : 1024 16B-chunks, 4 per thread
#pragma unroll
    for (int i = 0; i < 4; ++i) {
        const int ca  = tid + i * 256;
        const int row = ca >> 3;
        const int ch  = (ca & 7) << 2;               // k offset (floats)
        async_b128((unsigned int)(size_t)&shA[row * 32 + ch],
                   Ap + (long)(tm + row) * lda + k0 + ch);
    }
    // B panel : 512 16B-chunks, 2 per thread
#pragma unroll
    for (int i = 0; i < 2; ++i) {
        const int cb = tid + i * 256;
        if (WT) {
            const int row = cb >> 3;                 // n
            const int ch  = (cb & 7) << 2;           // k
            async_b128((unsigned int)(size_t)&shB[row * 32 + ch],
                       Bp + (long)(tn + row) * ldb + k0 + ch);
        } else {
            const int row = cb >> 4;                 // k
            const int ch  = (cb & 15) << 2;          // n
            async_b128((unsigned int)(size_t)&shB[row * 64 + ch],
                       Bp + (long)(k0 + row) * ldb + tn + ch);
        }
    }
}

// ---------------------------------------------------------------------------
// LDS-tiled, double-buffered WMMA GEMM. Block = 256 threads (8 waves) computes
// a 128x64 tile of C = alpha*(A@B) + bias(+bias2). Per K-step of 32:
//   - async-copies for step i+1 are issued BEFORE waiting on step i
//     (s_wait_asynccnt 6: async ops retire in order, so <=6 outstanding means
//      the 6 loads of the current buffer have landed) -> global latency is
//     hidden behind the 4 v_wmma_f32_16x16x32_f16 of the current step.
// A: [M,K] f32, row stride lda.
// B: WT==0 -> [K,N] (stride ldb), LDS layout [k][n] (32x64)
//    WT==1 -> [N,K] (stride ldb) i.e. B^T, LDS layout [n][k] (64x32)
// Batched via blockIdx.z: zo = z/bh_div, zi = z%bh_div, offsets zo*s?o+zi*s?i.
// Fragment layouts per CDNA5 ISA 7.12.2:
//   A f16 16x32: m=lane&15, k(e) = (e&7) + 8*(lane>>4) + 16*(e>>3)
//   B f16 32x16: n=lane&15, k(e) = (e&7) + 8*(lane>>4) + 16*(e>>3)
//   C/D f32 16x16: m = r + 8*(lane>>4), n = lane&15
// ---------------------------------------------------------------------------
template <int WT>
__global__ __launch_bounds__(256)
void gemm_tile(const float* __restrict__ A, const float* __restrict__ Bm,
               const float* __restrict__ bias, const float* __restrict__ bias2,
               float* __restrict__ C,
               int K, int lda, int ldb, int ldc,
               int bh_div, long sAo, long sAi, long sBo, long sBi,
               long sCo, long sCi, float alpha)
{
    __shared__ float shA[2][128 * 32];   // [buf][row][k]
    __shared__ float shB[2][64 * 32];    // WT==0: [k][n] 32x64 ; WT==1: [n][k] 64x32

    const int tid  = threadIdx.x;
    const int lane = tid & 31;
    const int w    = tid >> 5;           // wave id 0..7 -> rows 16w..16w+15
    const int hlf  = lane >> 4;
    const int r0   = lane & 15;
    const int tn   = blockIdx.x * 64;
    const int tm   = blockIdx.y * 128;
    const long zo  = blockIdx.z / bh_div;
    const long zi  = blockIdx.z % bh_div;

    const float* Ap = A  + zo * sAo + zi * sAi;
    const float* Bp = Bm + zo * sBo + zi * sBi;

    v8f acc[4] = {{}, {}, {}, {}};

    // prologue: fill buffer 0
    stage_tiles<WT>(shA[0], shB[0], Ap, Bp, tm, tn, 0, lda, ldb, tid);

    int buf = 0;
    for (int k0 = 0; k0 < K; k0 += 32, buf ^= 1) {
        const bool more = (k0 + 32 < K);
        if (more) {
            // prefetch next K-step into the other buffer (overlaps with WMMA)
            stage_tiles<WT>(shA[buf ^ 1], shB[buf ^ 1], Ap, Bp, tm, tn, k0 + 32,
                            lda, ldb, tid);
            asm volatile("s_wait_asynccnt 6" ::: "memory");
        } else {
            asm volatile("s_wait_asynccnt 0" ::: "memory");
        }
        __syncthreads();

        // ---- build A fragment (shared by the 4 column subtiles)
        const float* a0 = shA[buf];
        const float* b0 = shB[buf];
        v16h av;
#pragma unroll
        for (int e = 0; e < 16; ++e) {
            const int ka = (e & 7) + 8 * hlf + ((e >> 3) << 4);
            av[e] = (_Float16)a0[(w * 16 + r0) * 32 + ka];
        }
        // ---- 4 WMMAs across the 64-wide N tile
#pragma unroll
        for (int c = 0; c < 4; ++c) {
            v16h bv;
#pragma unroll
            for (int e = 0; e < 16; ++e) {
                const int kb = (e & 7) + 8 * hlf + ((e >> 3) << 4);
                bv[e] = WT ? (_Float16)b0[(c * 16 + r0) * 32 + kb]
                           : (_Float16)b0[kb * 64 + c * 16 + r0];
            }
            acc[c] = __builtin_amdgcn_wmma_f32_16x16x32_f16(false, av, false, bv,
                                                            (short)0, acc[c], false, false);
        }
        __syncthreads();
    }

    float* Cp = C + zo * sCo + zi * sCi;
#pragma unroll
    for (int c = 0; c < 4; ++c) {
        const int n = tn + c * 16 + r0;
        float bb = 0.f;
        if (bias)  bb += bias[n];
        if (bias2) bb += bias2[n];
#pragma unroll
        for (int r = 0; r < 8; ++r) {
            const int m = tm + w * 16 + r + 8 * hlf;
            Cp[(long)m * ldc + n] = acc[c][r] * alpha + bb;
        }
    }
}

// ---------------------------------------------------------------------------
// Embedding gather + LayerNorm. One block (256 threads) per token, 3 cols/thread.
// ---------------------------------------------------------------------------
__global__ void embed_ln(const int* __restrict__ idx, const float* __restrict__ we,
                         const float* __restrict__ pe, const float* __restrict__ te,
                         const float* __restrict__ g, const float* __restrict__ b,
                         float* __restrict__ x)
{
    __shared__ float sh[256];
    __shared__ float s_mean, s_rstd;
    const int t   = blockIdx.x;
    const int pos = t % SQ;
    const int wid = idx[t];
    const int tid = threadIdx.x;

    float vals[3];
    float sum = 0.f;
#pragma unroll
    for (int i = 0; i < 3; ++i) {
        const int c = tid + i * 256;
        float v = we[(long)wid * HD + c] + pe[(long)pos * HD + c] + te[c];
        vals[i] = v; sum += v;
    }
    sh[tid] = sum; __syncthreads();
    for (int s = 128; s > 0; s >>= 1) { if (tid < s) sh[tid] += sh[tid + s]; __syncthreads(); }
    if (tid == 0) s_mean = sh[0] / HD;
    __syncthreads();
    const float m = s_mean;
    float sq = 0.f;
#pragma unroll
    for (int i = 0; i < 3; ++i) { float d = vals[i] - m; sq += d * d; }
    __syncthreads();
    sh[tid] = sq; __syncthreads();
    for (int s = 128; s > 0; s >>= 1) { if (tid < s) sh[tid] += sh[tid + s]; __syncthreads(); }
    if (tid == 0) s_rstd = rsqrtf(sh[0] / HD + 1e-12f);
    __syncthreads();
    const float rs = s_rstd;
#pragma unroll
    for (int i = 0; i < 3; ++i) {
        const int c = tid + i * 256;
        x[(long)t * HD + c] = (vals[i] - m) * rs * g[c] + b[c];
    }
}

// ---------------------------------------------------------------------------
// x = LayerNorm(x + y). One block (256 threads) per token.
// ---------------------------------------------------------------------------
__global__ void add_ln(float* __restrict__ x, const float* __restrict__ y,
                       const float* __restrict__ g, const float* __restrict__ b)
{
    __shared__ float sh[256];
    __shared__ float s_mean, s_rstd;
    const int t   = blockIdx.x;
    const int tid = threadIdx.x;

    float vals[3];
    float sum = 0.f;
#pragma unroll
    for (int i = 0; i < 3; ++i) {
        const int c = tid + i * 256;
        float v = x[(long)t * HD + c] + y[(long)t * HD + c];
        vals[i] = v; sum += v;
    }
    sh[tid] = sum; __syncthreads();
    for (int s = 128; s > 0; s >>= 1) { if (tid < s) sh[tid] += sh[tid + s]; __syncthreads(); }
    if (tid == 0) s_mean = sh[0] / HD;
    __syncthreads();
    const float m = s_mean;
    float sq = 0.f;
#pragma unroll
    for (int i = 0; i < 3; ++i) { float d = vals[i] - m; sq += d * d; }
    __syncthreads();
    sh[tid] = sq; __syncthreads();
    for (int s = 128; s > 0; s >>= 1) { if (tid < s) sh[tid] += sh[tid + s]; __syncthreads(); }
    if (tid == 0) s_rstd = rsqrtf(sh[0] / HD + 1e-12f);
    __syncthreads();
    const float rs = s_rstd;
#pragma unroll
    for (int i = 0; i < 3; ++i) {
        const int c = tid + i * 256;
        x[(long)t * HD + c] = (vals[i] - m) * rs * g[c] + b[c];
    }
}

// ---------------------------------------------------------------------------
// Row softmax over SQ=128 wide rows. One block (128 threads) per row.
// ---------------------------------------------------------------------------
__global__ void softmax_rows(float* __restrict__ att)
{
    __shared__ float sh[128];
    float* p = att + (long)blockIdx.x * SQ;
    const int tid = threadIdx.x;
    float v = p[tid];
    sh[tid] = v; __syncthreads();
    for (int s = 64; s > 0; s >>= 1) { if (tid < s) sh[tid] = fmaxf(sh[tid], sh[tid + s]); __syncthreads(); }
    const float mx = sh[0]; __syncthreads();
    const float e = expf(v - mx);
    sh[tid] = e; __syncthreads();
    for (int s = 64; s > 0; s >>= 1) { if (tid < s) sh[tid] += sh[tid + s]; __syncthreads(); }
    p[tid] = e / sh[0];
}

// ---------------------------------------------------------------------------
// Exact GELU (erf form), in place.
// ---------------------------------------------------------------------------
__global__ void gelu_k(float* __restrict__ y, long n)
{
    const long i = (long)blockIdx.x * blockDim.x + threadIdx.x;
    if (i < n) {
        const float v = y[i];
        y[i] = 0.5f * v * (1.f + erff(v * 0.70710678118654752f));
    }
}

// ---------------------------------------------------------------------------
// LSTM recurrent scan. Input gates gx = x @ Wih^T + (bih+bhh) precomputed by
// WMMA GEMM. One block per batch; 1024 threads = one per gate (4*LH).
// Gate order (PyTorch): i, f, g, o.
// ---------------------------------------------------------------------------
__global__ __launch_bounds__(1024)
void lstm_scan(const float* __restrict__ gx, const float* __restrict__ Whh,
               float* __restrict__ hseq)
{
    __shared__ float h[LH], c[LH], gates[G4];
    const int b   = blockIdx.x;
    const int tid = threadIdx.x;
    if (tid < LH) { h[tid] = 0.f; c[tid] = 0.f; }
    __syncthreads();
    for (int t = 0; t < SQ; ++t) {
        float acc = gx[(long)(b * SQ + t) * G4 + tid];
        const float* w = Whh + (long)tid * LH;
#pragma unroll 4
        for (int j = 0; j < LH; ++j) acc += w[j] * h[j];
        gates[tid] = acc;
        __syncthreads();
        if (tid < LH) {
            const float ig = 1.f / (1.f + expf(-gates[tid]));
            const float fg = 1.f / (1.f + expf(-gates[LH + tid]));
            const float gg = tanhf(gates[2 * LH + tid]);
            const float og = 1.f / (1.f + expf(-gates[3 * LH + tid]));
            const float cc = fg * c[tid] + ig * gg;
            c[tid] = cc;
            const float hh = og * tanhf(cc);
            h[tid] = hh;
            hseq[(long)(b * SQ + t) * LH + tid] = hh;
        }
        __syncthreads();
    }
}

// ---------------------------------------------------------------------------
// Classifier: emis = h @ W^T + b  ([NT,LH] x [NC,LH]^T -> [NT,NC])
// ---------------------------------------------------------------------------
__global__ void cls_k(const float* __restrict__ h2, const float* __restrict__ W,
                      const float* __restrict__ b, float* __restrict__ emis)
{
    const int i = blockIdx.x * blockDim.x + threadIdx.x;
    if (i >= NT * NC) return;
    const int t = i / NC, cc = i % NC;
    float acc = b[cc];
    const float* hp = h2 + (long)t * LH;
    const float* wp = W  + (long)cc * LH;
    for (int j = 0; j < LH; ++j) acc += hp[j] * wp[j];
    emis[i] = acc;
}

// ---------------------------------------------------------------------------
// CRF negative log-likelihood (sum reduction). One block of 128 threads.
// ---------------------------------------------------------------------------
__global__ void crf_k(const float* __restrict__ emis, const int* __restrict__ label,
                      const float* __restrict__ start, const float* __restrict__ endv,
                      const float* __restrict__ trans, float* __restrict__ out)
{
    __shared__ float al[NB * NC], al2[NB * NC], numer[NB], lz[NB];
    const int tid = threadIdx.x;

    if (tid < NB) {
        const int b = tid;
        const int* lb = label + b * SQ;
        float num = start[lb[0]] + endv[lb[SQ - 1]];
        for (int t = 0; t < SQ; ++t)     num += emis[(long)(b * SQ + t) * NC + lb[t]];
        for (int t = 0; t < SQ - 1; ++t) num += trans[lb[t] * NC + lb[t + 1]];
        numer[b] = num;
    }
    if (tid < NB * NC) {
        const int b = tid / NC, j = tid % NC;
        al[tid] = start[j] + emis[(long)(b * SQ) * NC + j];
    }
    __syncthreads();
    for (int t = 1; t < SQ; ++t) {
        if (tid < NB * NC) {
            const int b = tid / NC, j = tid % NC;
            float mx = -1e30f;
            for (int i = 0; i < NC; ++i) mx = fmaxf(mx, al[b * NC + i] + trans[i * NC + j]);
            float s = 0.f;
            for (int i = 0; i < NC; ++i) s += expf(al[b * NC + i] + trans[i * NC + j] - mx);
            al2[tid] = mx + logf(s) + emis[(long)(b * SQ + t) * NC + j];
        }
        __syncthreads();
        if (tid < NB * NC) al[tid] = al2[tid];
        __syncthreads();
    }
    if (tid < NB) {
        const int b = tid;
        float mx = -1e30f;
        for (int j = 0; j < NC; ++j) mx = fmaxf(mx, al[b * NC + j] + endv[j]);
        float s = 0.f;
        for (int j = 0; j < NC; ++j) s += expf(al[b * NC + j] + endv[j] - mx);
        lz[b] = mx + logf(s);
    }
    __syncthreads();
    if (tid == 0) {
        float loss = 0.f;
        for (int b = 0; b < NB; ++b) loss += lz[b] - numer[b];
        out[0] = loss;
    }
}

// ---------------------------------------------------------------------------
extern "C" void kernel_launch(void* const* d_in, const int* in_sizes, int n_in,
                              void* d_out, int out_size, void* d_ws, size_t ws_size,
                              hipStream_t stream)
{
    (void)in_sizes; (void)n_in; (void)out_size; (void)ws_size;

    const int*   batch_index = (const int*)  d_in[0];
    const int*   batch_label = (const int*)  d_in[1];
    const float* word_emb    = (const float*)d_in[2];
    const float* pos_emb     = (const float*)d_in[3];
    const float* type_emb    = (const float*)d_in[4];
    const float* emb_g       = (const float*)d_in[5];
    const float* emb_b       = (const float*)d_in[6];
    const float* Wq = (const float*)d_in[7];  const float* bq = (const float*)d_in[8];
    const float* Wk = (const float*)d_in[9];  const float* bk = (const float*)d_in[10];
    const float* Wv = (const float*)d_in[11]; const float* bv = (const float*)d_in[12];
    const float* Wo = (const float*)d_in[13]; const float* bo = (const float*)d_in[14];
    const float* ln1g = (const float*)d_in[15]; const float* ln1b = (const float*)d_in[16];
    const float* W1 = (const float*)d_in[17]; const float* b1 = (const float*)d_in[18];
    const float* W2 = (const float*)d_in[19]; const float* b2 = (const float*)d_in[20];
    const float* ln2g = (const float*)d_in[21]; const float* ln2b = (const float*)d_in[22];
    const float* Wih0 = (const float*)d_in[23]; const float* Whh0 = (const float*)d_in[24];
    const float* bih0 = (const float*)d_in[25]; const float* bhh0 = (const float*)d_in[26];
    const float* Wih1 = (const float*)d_in[27]; const float* Whh1 = (const float*)d_in[28];
    const float* bih1 = (const float*)d_in[29]; const float* bhh1 = (const float*)d_in[30];
    const float* clsW = (const float*)d_in[31]; const float* clsB = (const float*)d_in[32];
    const float* crfS = (const float*)d_in[33]; const float* crfE = (const float*)d_in[34];
    const float* crfT = (const float*)d_in[35];
    float* out = (float*)d_out;

    // workspace carve-up (floats)
    float* ws  = (float*)d_ws;
    float* x   = ws;
    float* qb  = x   + (long)NT * HD;
    float* kb  = qb  + (long)NT * HD;
    float* vb  = kb  + (long)NT * HD;
    float* ctx = vb  + (long)NT * HD;
    float* att = ctx + (long)NT * HD;
    float* ffb = att + (long)NB * NHEAD * SQ * SQ;
    float* gx  = ffb + (long)NT * FFD;
    float* h1  = gx  + (long)NT * G4;
    float* h2  = h1  + (long)NT * LH;
    float* emi = h2  + (long)NT * LH;

    auto G = [&](const float* A, const float* Bm, const float* bias, const float* bias2,
                 float* C, int M, int N, int K, int lda, int ldb, int ldc,
                 int Z, int bh_div, long sAo, long sAi, long sBo, long sBi,
                 long sCo, long sCi, float alpha, bool wt) {
        dim3 grid(N / 64, M / 128, Z);
        if (wt) gemm_tile<1><<<grid, 256, 0, stream>>>(A, Bm, bias, bias2, C, K, lda, ldb, ldc,
                                                       bh_div, sAo, sAi, sBo, sBi, sCo, sCi, alpha);
        else    gemm_tile<0><<<grid, 256, 0, stream>>>(A, Bm, bias, bias2, C, K, lda, ldb, ldc,
                                                       bh_div, sAo, sAi, sBo, sBi, sCo, sCi, alpha);
    };

    // Embeddings + LN
    embed_ln<<<NT, 256, 0, stream>>>(batch_index, word_emb, pos_emb, type_emb, emb_g, emb_b, x);

    const float scale = 0.125f;  // 1/sqrt(64)
    for (int i = 0; i < NLAY; ++i) {
        const float* Wqi = Wq + (long)i * HD * HD; const float* bqi = bq + (long)i * HD;
        const float* Wki = Wk + (long)i * HD * HD; const float* bki = bk + (long)i * HD;
        const float* Wvi = Wv + (long)i * HD * HD; const float* bvi = bv + (long)i * HD;
        const float* Woi = Wo + (long)i * HD * HD; const float* boi = bo + (long)i * HD;
        const float* W1i = W1 + (long)i * HD * FFD; const float* b1i = b1 + (long)i * FFD;
        const float* W2i = W2 + (long)i * FFD * HD; const float* b2i = b2 + (long)i * HD;

        // QKV projections
        G(x, Wqi, bqi, nullptr, qb, NT, HD, HD, HD, HD, HD, 1, 1, 0,0,0,0,0,0, 1.f, false);
        G(x, Wki, bki, nullptr, kb, NT, HD, HD, HD, HD, HD, 1, 1, 0,0,0,0,0,0, 1.f, false);
        G(x, Wvi, bvi, nullptr, vb, NT, HD, HD, HD, HD, HD, 1, 1, 0,0,0,0,0,0, 1.f, false);
        // scores[b,h] = (Q K^T) * scale : per-(b,h) GEMM, M=N=128, K=64 (B^T case)
        G(qb, kb, nullptr, nullptr, att, SQ, SQ, DHEAD, HD, HD, SQ,
          NB * NHEAD, NHEAD,
          (long)SQ * HD, DHEAD, (long)SQ * HD, DHEAD,
          (long)NHEAD * SQ * SQ, (long)SQ * SQ, scale, true);
        softmax_rows<<<NB * NHEAD * SQ, 128, 0, stream>>>(att);
        // ctx[b,h] = att @ V : per-(b,h), M=128, N=64, K=128
        G(att, vb, nullptr, nullptr, ctx, SQ, DHEAD, SQ, SQ, HD, HD,
          NB * NHEAD, NHEAD,
          (long)NHEAD * SQ * SQ, (long)SQ * SQ, (long)SQ * HD, DHEAD,
          (long)SQ * HD, DHEAD, 1.f, false);
        // output projection -> qb (reuse), then x = LN(x + proj)
        G(ctx, Woi, boi, nullptr, qb, NT, HD, HD, HD, HD, HD, 1, 1, 0,0,0,0,0,0, 1.f, false);
        add_ln<<<NT, 256, 0, stream>>>(x, qb, ln1g + (long)i * HD, ln1b + (long)i * HD);
        // FFN
        G(x, W1i, b1i, nullptr, ffb, NT, FFD, HD, HD, FFD, FFD, 1, 1, 0,0,0,0,0,0, 1.f, false);
        gelu_k<<<(int)(((long)NT * FFD) / 256), 256, 0, stream>>>(ffb, (long)NT * FFD);
        G(ffb, W2i, b2i, nullptr, qb, NT, HD, FFD, FFD, HD, HD, 1, 1, 0,0,0,0,0,0, 1.f, false);
        add_ln<<<NT, 256, 0, stream>>>(x, qb, ln2g + (long)i * HD, ln2b + (long)i * HD);
    }

    // LSTM layer 0: gates_x = x @ Wih0^T + (bih0 + bhh0)   (WMMA, B^T case)
    G(x, Wih0, bih0, bhh0, gx, NT, G4, HD, HD, HD, G4, 1, 1, 0,0,0,0,0,0, 1.f, true);
    lstm_scan<<<NB, 1024, 0, stream>>>(gx, Whh0, h1);
    // LSTM layer 1
    G(h1, Wih1, bih1, bhh1, gx, NT, G4, LH, LH, LH, G4, 1, 1, 0,0,0,0,0,0, 1.f, true);
    lstm_scan<<<NB, 1024, 0, stream>>>(gx, Whh1, h2);

    // Classifier + CRF loss
    cls_k<<<(NT * NC + 255) / 256, 256, 0, stream>>>(h2, clsW, clsB, emi);
    crf_k<<<1, 128, 0, stream>>>(emi, batch_label, crfS, crfE, crfT, out);
}
